// CategoryBranch_40278203302018
// MI455X (gfx1250) — matrix operand order repaired and checked by reading, not compiled, not run on hardware
//
#include <hip/hip_runtime.h>
#include <hip/hip_bf16.h>

// ---- problem constants (from reference) ----
#define B_       32
#define L_       64      // Hc (attention positions)
#define F_       8192    // C*Wc = 64*128  (GRU input features)
#define H_       512     // DDEC
#define G_       1536    // 3*H
#define T_       64      // output_len
#define NCLS_    17
#define NBINS_   128
#define NFRAMES_ 256
#define NFILT_   64
#define WC_      128     // Wc
#define FCHUNKS_ 4       // attn F-split (gridDim.y)

typedef __bf16 bf16;
typedef __attribute__((ext_vector_type(16))) __bf16 v16bf;
typedef __attribute__((ext_vector_type(8)))  __bf16 v8bf;
typedef __attribute__((ext_vector_type(8)))  float  v8f;

// ---------------- one-time kernels ----------------

// Conv2d stride(2,2) pad(1,1), CIN=1, then transpose(1,2)+reshape into
// context[b][l][c*WC + w] stored directly in bf16.
__global__ void conv_to_ctx(const float* __restrict__ x,
                            const float* __restrict__ w,
                            bf16* __restrict__ ctx) {
  int idx = blockIdx.x * blockDim.x + threadIdx.x;   // B*L*NFILT*WC = 16.7M
  int wc = idx & (WC_ - 1);
  int c  = (idx >> 7) & 63;
  int l  = (idx >> 13) & 63;
  int b  = idx >> 19;
  const float* wk = w + c * 9;
  float acc = 0.f;
#pragma unroll
  for (int kh = 0; kh < 3; ++kh) {
    int ih = 2 * l - 1 + kh;
    if (ih < 0 || ih >= NBINS_) continue;
#pragma unroll
    for (int kw = 0; kw < 3; ++kw) {
      int iw = 2 * wc - 1 + kw;
      if (iw < 0 || iw >= NFRAMES_) continue;
      acc += x[((size_t)b * NBINS_ + ih) * NFRAMES_ + iw] * wk[kh * 3 + kw];
    }
  }
  ctx[((size_t)(b * L_ + l)) * F_ + c * WC_ + wc] = (bf16)acc;
}

__global__ void f32_to_bf16(const float* __restrict__ src,
                            bf16* __restrict__ dst, int n) {
  int i = blockIdx.x * blockDim.x + threadIdx.x;
  if (i < n) dst[i] = (bf16)src[i];
}

__global__ void init_state(float* __restrict__ hidden, bf16* __restrict__ hbf,
                           float* __restrict__ kappa0) {
  int i = blockIdx.x * blockDim.x + threadIdx.x;
  if (i < B_ * H_) { hidden[i] = 0.f; hbf[i] = (bf16)0.f; }
  if (i < B_) kappa0[i] = 0.f;
}

// ---------------- per-step kernels ----------------

// Grid (B, FCHUNKS). Every block recomputes kappa/beta for its batch row
// (reads double-buffered kappa_in, only y==0 writes kappa_out and ws), then
// handles an F_/FCHUNKS_ slice of attended = sum_l w[l]*context[b,l,:].
__global__ void __launch_bounds__(256)
attn_step(const float* __restrict__ hidden,
          const float* __restrict__ kappa_in, float* __restrict__ kappa_out,
          const float* __restrict__ kappa_w, const float* __restrict__ beta_w,
          const bf16* __restrict__ ctx, bf16* __restrict__ att,
          float* __restrict__ ws_out, int t) {
  int b = blockIdx.x, chunk = blockIdx.y, tid = threadIdx.x;
  __shared__ float redk[256], redb[256];
  __shared__ float s_w[L_];
  __shared__ float s_kb[2];
  float pk = 0.f, pb = 0.f;
  for (int d = tid; d < H_; d += 256) {
    float h = hidden[b * H_ + d];
    pk += h * kappa_w[d];
    pb += h * beta_w[d];
  }
  redk[tid] = pk; redb[tid] = pb;
  __syncthreads();
  for (int s = 128; s > 0; s >>= 1) {
    if (tid < s) { redk[tid] += redk[tid + s]; redb[tid] += redb[tid + s]; }
    __syncthreads();
  }
  if (tid == 0) {
    float kn = kappa_in[b] + redk[0];
    if (chunk == 0) kappa_out[b] = kn;   // persist for next step (double buffer)
    s_kb[0] = kn;
    s_kb[1] = expf(redb[0]);             // beta
  }
  __syncthreads();
  if (tid < L_) {
    float d  = s_kb[0] - (float)tid;
    float wv = expf(-s_kb[1] * d * d);
    s_w[tid] = wv;
    if (chunk == 0)
      ws_out[((size_t)b * T_ + t) * L_ + tid] = wv;
  }
  __syncthreads();
  const bf16* cb = ctx + (size_t)b * L_ * F_;
  const int fbeg = chunk * (F_ / FCHUNKS_);
  const int fend = fbeg + (F_ / FCHUNKS_);
  for (int f = fbeg + tid; f < fend; f += 256) {
    float acc = 0.f;
#pragma unroll 8
    for (int l = 0; l < L_; ++l)
      acc += s_w[l] * (float)cb[(size_t)l * F_ + f];
    att[(size_t)b * F_ + f] = (bf16)acc;
  }
}

// C[m][n] = sum_k A[m][k] * B[n][k]   (fp32 accum, bf16 inputs)
// One wave per 16(M) x 32(N) tile: two adjacent n-tiles share the A fragment,
// giving two independent WMMA accumulation chains (XDL pipelining) and halving
// A-panel L2 traffic. WMMA VGPR layouts used here (A/B share a contiguous k order; C/D per CDNA5 ISA 7.12.2):
//  A 16x32: lane m=lane&15; hi=lane>>4; elems 0..7 -> K=k0+16*hi+j,
//           elems 8..15 -> K=k0+16*hi+8+j  (two contiguous b128 chunks)
//  B 32x16: lane n=lane&15; elems j -> K=k0+16*hi+j (one contiguous 32B chunk)
//  C/D:     m = 8*hi + vgpr, n = lane&15
__global__ void __launch_bounds__(256)
wmma_gemm_bf16(const bf16* __restrict__ A, int lda,
               const bf16* __restrict__ B, int ldb,
               float* __restrict__ C, int ldc,
               int pairs_n, int total_pairs, int K) {
  int wid  = (int)((blockIdx.x * blockDim.x + threadIdx.x) >> 5);
  if (wid >= total_pairs) return;
  int lane = threadIdx.x & 31;
  int tm = wid / pairs_n, pn = wid % pairs_n;
  int hi  = lane >> 4;
  int lid = lane & 15;
  const bf16* arow  = A + (size_t)(tm * 16 + lid) * lda + hi * 16;
  const bf16* brow0 = B + (size_t)(pn * 32 + lid) * ldb + hi * 16;
  const bf16* brow1 = brow0 + (size_t)16 * ldb;
  v8f acc0 = {}, acc1 = {};
#pragma unroll 4
  for (int k0 = 0; k0 < K; k0 += 32) {
    v8bf a_lo = *(const v8bf*)(arow + k0);
    v8bf a_hi = *(const v8bf*)(arow + k0 + 8);
    v16bf a = __builtin_shufflevector(a_lo, a_hi,
                                      0, 1, 2, 3, 4, 5, 6, 7,
                                      8, 9, 10, 11, 12, 13, 14, 15);
    v16bf bm0 = *(const v16bf*)(brow0 + k0);
    v16bf bm1 = *(const v16bf*)(brow1 + k0);
    acc0 = __builtin_amdgcn_wmma_f32_16x16x32_bf16(
        false, a, false, bm0, (short)0, acc0, false, false);
    acc1 = __builtin_amdgcn_wmma_f32_16x16x32_bf16(
        false, a, false, bm1, (short)0, acc1, false, false);
  }
  float* crow0 = C + (size_t)(tm * 16 + hi * 8) * ldc + pn * 32 + lid;
  float* crow1 = crow0 + 16;
#pragma unroll
  for (int v = 0; v < 8; ++v) {
    crow0[(size_t)v * ldc] = acc0[v];
    crow1[(size_t)v * ldc] = acc1[v];
  }
}

// GRU gating (torch order r,z,n; no bias) + classifier head, one block/batch.
__global__ void __launch_bounds__(256)
gru_out_step(const float* __restrict__ gx, const float* __restrict__ gh,
             float* __restrict__ hidden, bf16* __restrict__ hidden_bf,
             const float* __restrict__ out_w, const float* __restrict__ out_b,
             float* __restrict__ outs, int t) {
  int b = blockIdx.x, tid = threadIdx.x;
  __shared__ float hn[H_];
  __shared__ float part[NCLS_ * 8];
  const float* gxb = gx + (size_t)b * G_;
  const float* ghb = gh + (size_t)b * G_;
  for (int d = tid; d < H_; d += 256) {
    float r = 1.f / (1.f + expf(-(gxb[d] + ghb[d])));
    float z = 1.f / (1.f + expf(-(gxb[H_ + d] + ghb[H_ + d])));
    float n = tanhf(gxb[2 * H_ + d] + r * ghb[2 * H_ + d]);
    float h = (1.f - z) * n + z * hidden[(size_t)b * H_ + d];
    hn[d] = h;
    hidden[(size_t)b * H_ + d] = h;
    hidden_bf[(size_t)b * H_ + d] = (bf16)h;
  }
  __syncthreads();
  if (tid < NCLS_ * 8) {
    int c = tid >> 3, l8 = tid & 7;
    float acc = 0.f;
    for (int d = l8; d < H_; d += 8) acc += hn[d] * out_w[c * H_ + d];
    part[tid] = acc;
  }
  __syncthreads();
  if (tid < NCLS_) {
    float s = out_b[tid];
#pragma unroll
    for (int j = 0; j < 8; ++j) s += part[tid * 8 + j];
    outs[((size_t)b * T_ + t) * NCLS_ + tid] = s;
  }
}

// ---------------- host launcher ----------------

extern "C" void kernel_launch(void* const* d_in, const int* in_sizes, int n_in,
                              void* d_out, int out_size, void* d_ws, size_t ws_size,
                              hipStream_t stream) {
  const float* x       = (const float*)d_in[0];
  // d_in[1] = output_len (device int) — fixed at T_=64 per problem constants
  const float* conv_w  = (const float*)d_in[2];
  const float* kappa_w = (const float*)d_in[3];
  const float* beta_w  = (const float*)d_in[4];
  const float* w_ih    = (const float*)d_in[5];
  const float* w_hh    = (const float*)d_in[6];
  const float* out_w   = (const float*)d_in[7];
  const float* out_b   = (const float*)d_in[8];

  // workspace carve-up (256B aligned regions)
  char* ws = (char*)d_ws;
  size_t off = 0;
  auto carve = [&](size_t bytes) -> char* {
    char* p = ws + off;
    off = (off + bytes + 255) & ~(size_t)255;
    return p;
  };
  bf16*  ctx  = (bf16*)carve((size_t)B_ * L_ * F_ * sizeof(bf16)); // 32 MB
  bf16*  wihb = (bf16*)carve((size_t)G_ * F_ * sizeof(bf16));      // 24 MB
  bf16*  whhb = (bf16*)carve((size_t)G_ * H_ * sizeof(bf16));
  bf16*  attb = (bf16*)carve((size_t)B_ * F_ * sizeof(bf16));
  float* hid  = (float*)carve((size_t)B_ * H_ * sizeof(float));
  bf16*  hbf  = (bf16*)carve((size_t)B_ * H_ * sizeof(bf16));
  float* kap0 = (float*)carve((size_t)B_ * sizeof(float));
  float* kap1 = (float*)carve((size_t)B_ * sizeof(float));
  float* gxp  = (float*)carve((size_t)B_ * G_ * sizeof(float));
  float* ghp  = (float*)carve((size_t)B_ * G_ * sizeof(float));
  (void)ws_size; (void)in_sizes; (void)n_in; (void)out_size;

  float* outs  = (float*)d_out;                           // (B,T,17)
  float* wsout = (float*)d_out + (size_t)B_ * T_ * NCLS_; // (B,T,64)

  // one-time prep (per launch, deterministic)
  f32_to_bf16<<<(G_ * F_ + 255) / 256, 256, 0, stream>>>(w_ih, wihb, G_ * F_);
  f32_to_bf16<<<(G_ * H_ + 255) / 256, 256, 0, stream>>>(w_hh, whhb, G_ * H_);
  conv_to_ctx<<<(B_ * L_ * NFILT_ * WC_) / 256, 256, 0, stream>>>(x, conv_w, ctx);
  init_state<<<(B_ * H_ + 255) / 256, 256, 0, stream>>>(hid, hbf, kap0);

  const int pairs_n     = (G_ / 16) / 2;   // 48 pairs of n-tiles
  const int total_pairs = 2 * pairs_n;     // M=32 -> 2 m-tiles; 96 waves
  const int gemm_blocks = (total_pairs * 32 + 255) / 256; // 12 blocks, 8 waves ea

  for (int t = 0; t < T_; ++t) {
    float* kin  = (t & 1) ? kap1 : kap0;
    float* kout = (t & 1) ? kap0 : kap1;
    attn_step<<<dim3(B_, FCHUNKS_), 256, 0, stream>>>(
        hid, kin, kout, kappa_w, beta_w, ctx, attb, wsout, t);
    // gx = attended @ W_ih^T   (K = 8192)
    wmma_gemm_bf16<<<gemm_blocks, 256, 0, stream>>>(attb, F_, wihb, F_,
                                                    gxp, G_, pairs_n,
                                                    total_pairs, F_);
    // gh = hidden @ W_hh^T     (K = 512)
    wmma_gemm_bf16<<<gemm_blocks, 256, 0, stream>>>(hbf, H_, whhb, H_,
                                                    ghp, G_, pairs_n,
                                                    total_pairs, H_);
    gru_out_step<<<B_, 256, 0, stream>>>(gxp, ghp, hid, hbf, out_w, out_b,
                                         outs, t);
  }
}
